// ViT_UCB_Pruning_60481729462844
// MI455X (gfx1250) — matrix operand order repaired and checked by reading, not compiled, not run on hardware
//
#include <hip/hip_runtime.h>
#include <hip/hip_bf16.h>
#include <cstdint>
#include <cstddef>

// ---------------- types / constants ----------------
typedef _Float16 f16_t;
typedef __attribute__((ext_vector_type(16))) _Float16 v16h;
typedef __attribute__((ext_vector_type(8)))  float    v8f;
typedef __attribute__((ext_vector_type(4)))  unsigned int u32x4;
typedef __attribute__((ext_vector_type(8)))  unsigned int u32x8;

union HV { uint4 q[2]; v16h v; };  // 32B fragment assembly from two b128 loads

#define B_   32
#define N_   197
#define C_   1024
#define H_   16
#define DH_  64
#define M_   (B_ * N_)   // 6304 rows (divisible by 32)
#define NP_  208         // padded prob row stride (13*16)
#define NPL_ 224         // LDS/vT padded row (multiple of 32 for K-steps)
#define KKEEP 98         // max(1, int(196*0.5))

static __device__ __forceinline__ v8f wmma_f16(v16h a, v16h b, v8f c) {
  return __builtin_amdgcn_wmma_f32_16x16x32_f16(false, a, false, b, (short)0, c,
                                                false, false);
}

// ---------------- Tensor Data Mover: 2D tile (16-bit elems) -> LDS ----------
// D# per CDNA5 ISA. Tile rows land contiguously
// in LDS (no padding). Issued once per calling wave; tracked by TENSORcnt.
static __device__ __forceinline__ void tdm_load_2d(
    uint32_t lds_addr, const void* gptr, uint32_t tensor_d0, uint32_t tensor_d1,
    uint32_t stride_elems, uint32_t tile_d0, uint32_t tile_d1) {
  uint64_t ga = (uint64_t)(uintptr_t)gptr;
  // group0: count=1 | lds_addr | global_addr[56:0] | type=2
  u32x4 g0 = {1u, lds_addr, (uint32_t)ga,
              (uint32_t)((ga >> 32) & 0x01FFFFFFu) | (2u << 30)};
  // group1: data_size=1 (2B); tensor_dim0/1; tile_dim0/1; dim0 stride
  u32x8 g1 = {
      (1u << 16),                                          // data_size=2B
      (tensor_d0 & 0xFFFFu) << 16,                         // tensor_dim0 lo16
      (tensor_d0 >> 16) | ((tensor_d1 & 0xFFFFu) << 16),   // td0 hi | td1 lo
      (tensor_d1 >> 16) | (tile_d0 << 16),                 // td1 hi | tile_d0
      tile_d1,                                             // tile_d1 | tile_d2=0
      stride_elems,                                        // dim0 stride lo32
      0u,                                                  // stride hi | d1str lo
      0u};
  asm volatile("tensor_load_to_lds %0, %1" :: "s"(g0), "s"(g1) : "memory");
}

// ---------------- fp32 -> f16 convert (row-major kept) ----------------
__global__ void cvt_f32_f16(const float* __restrict__ src,
                            f16_t* __restrict__ dst, int n4) {
  int i = blockIdx.x * 256 + threadIdx.x;
  if (i < n4) {
    float4 v = ((const float4*)src)[i];
    f16_t o[4] = {(f16_t)v.x, (f16_t)v.y, (f16_t)v.z, (f16_t)v.w};
    *(uint64_t*)&dst[(size_t)i * 4] = *(uint64_t*)o;
  }
}

// ---------------- fp32 [K][Nc] -> f16 transposed [Nc][K] (32x32 LDS tiles) ---
__global__ void __launch_bounds__(256) cvt_transpose(
    const float* __restrict__ src, f16_t* __restrict__ dst, int K, int Nc) {
  __shared__ float tile[32][33];
  const int kt = blockIdx.y * 32, nt = blockIdx.x * 32;
  const int r = threadIdx.x >> 3;        // 0..31
  const int c4 = (threadIdx.x & 7) * 4;  // 0,4,..,28
  float4 v = *(const float4*)(src + (size_t)(kt + r) * Nc + nt + c4);
  tile[r][c4 + 0] = v.x; tile[r][c4 + 1] = v.y;
  tile[r][c4 + 2] = v.z; tile[r][c4 + 3] = v.w;
  __syncthreads();
  f16_t o[4] = {(f16_t)tile[c4 + 0][r], (f16_t)tile[c4 + 1][r],
                (f16_t)tile[c4 + 2][r], (f16_t)tile[c4 + 3][r]};
  *(uint64_t*)(dst + (size_t)(nt + r) * K + kt + c4) = *(uint64_t*)o;
}

// ---------------- per-(b,h) V transpose: QKV f16 -> vT[d][j] (padded) -------
__global__ void __launch_bounds__(256) transpose_v(
    const f16_t* __restrict__ QKVh, f16_t* __restrict__ vT) {
  __shared__ f16_t t[32][40];
  const int bh = blockIdx.z;
  const int b = bh >> 4, h = bh & 15;
  const int j0 = blockIdx.x * 32, d0 = blockIdx.y * 32;
  const int r = threadIdx.x >> 3, c4 = (threadIdx.x & 7) * 4;
  const int j = j0 + r;
  uint64_t val = 0;
  if (j < N_)
    val = *(const uint64_t*)(QKVh + (size_t)(b * N_ + j) * (3 * C_) + 2 * C_ +
                             h * DH_ + d0 + c4);
  *(uint64_t*)&t[r][c4] = val;
  __syncthreads();
  f16_t o[4] = {t[c4 + 0][r], t[c4 + 1][r], t[c4 + 2][r], t[c4 + 3][r]};
  *(uint64_t*)(vT + (size_t)(bh * DH_ + d0 + r) * NPL_ + j0 + c4) =
      *(uint64_t*)o;
}

// ---------------- generic WMMA GEMM: TDM double-buffered LDS staging -------
// C[M,N] = A[M,K] * Bt[N,K]^T (+ bias). M%32==0, N%256==0, K%32==0.
// 256 threads = 8 waves; block tile 32x256; wave owns 32x32 (2x2 WMMAs).
// Wave 0 drives the Tensor Data Mover; tile t+1 DMA overlaps tile t compute.
__global__ void __launch_bounds__(256) gemm_wmma(
    const f16_t* __restrict__ A, const f16_t* __restrict__ Bt,
    f16_t* __restrict__ Cf16, float* __restrict__ Cf32,
    const float* __restrict__ bias, int M, int N, int K) {
  __shared__ f16_t sA[2][32][32];     // TDM tiles: 32 rows x 32 halves each
  __shared__ f16_t sBt[2][256][32];   // TDM tiles: 256 n-rows x 32 halves each
  const int tid = threadIdx.x;
  const int lane = tid & 31, w = tid >> 5;
  const int m0 = blockIdx.y * 32;
  const int n0 = blockIdx.x * 256;
  const int nk = K >> 5;  // K / 32 steps
  v8f acc[2][2] = {};

  if (tid < 32) {  // prologue: DMA tile 0 into buffer 0
    tdm_load_2d((uint32_t)(uintptr_t)&sA[0][0][0], A + (size_t)m0 * K,
                K, M, K, 32, 32);
    tdm_load_2d((uint32_t)(uintptr_t)&sBt[0][0][0], Bt + (size_t)n0 * K,
                K, N, K, 32, 256);
  }

  for (int t = 0; t < nk; ++t) {
    const int buf = t & 1;
    if (tid < 32) {
      if (t + 1 < nk) {  // issue tile t+1 into the other buffer, then wait
        const int kk1 = (t + 1) << 5;
        tdm_load_2d((uint32_t)(uintptr_t)&sA[buf ^ 1][0][0],
                    A + (size_t)m0 * K + kk1, K, M, K, 32, 32);
        tdm_load_2d((uint32_t)(uintptr_t)&sBt[buf ^ 1][0][0],
                    Bt + (size_t)n0 * K + kk1, K, N, K, 32, 256);
        if (t + 2 < nk) {  // L2 hint for tile t+2 (global_prefetch_b8)
          __builtin_prefetch(A + (size_t)m0 * K + kk1 + 32, 0, 1);
          __builtin_prefetch(Bt + (size_t)n0 * K + kk1 + 32, 0, 1);
        }
        // <=2 outstanding: only the just-issued pair pending => tile t done
        __builtin_amdgcn_s_wait_tensorcnt(2);
      } else {
        __builtin_amdgcn_s_wait_tensorcnt(0);
      }
    }
    __syncthreads();

    v16h a[2], b[2];
#pragma unroll
    for (int mt = 0; mt < 2; ++mt) {  // A frag: 2x b128, K-contiguous chunks
      const f16_t* ap = &sA[buf][mt * 16 + (lane & 15)][(lane >> 4) << 3];
      HV tt;
      tt.q[0] = *(const uint4*)ap;         // K = off .. off+7
      tt.q[1] = *(const uint4*)(ap + 16);  // K = off+16 .. off+23
      a[mt] = tt.v;
    }
#pragma unroll
    for (int nt = 0; nt < 2; ++nt) {  // B frag: 2x b128, same k order as A
      const f16_t* bp =
          &sBt[buf][w * 32 + nt * 16 + (lane & 15)][(lane >> 4) << 3];
      HV tt;
      tt.q[0] = *(const uint4*)bp;
      tt.q[1] = *(const uint4*)(bp + 16);
      b[nt] = tt.v;
    }
#pragma unroll
    for (int mt = 0; mt < 2; ++mt)
#pragma unroll
      for (int nt = 0; nt < 2; ++nt)
        acc[mt][nt] = wmma_f16(a[mt], b[nt], acc[mt][nt]);
    __syncthreads();
  }

#pragma unroll
  for (int mt = 0; mt < 2; ++mt)
#pragma unroll
    for (int nt = 0; nt < 2; ++nt)
#pragma unroll
      for (int r = 0; r < 8; ++r) {
        int row = m0 + mt * 16 + r + ((lane >> 4) << 3);
        int col = n0 + w * 32 + nt * 16 + (lane & 15);
        float v = acc[mt][nt][r];
        if (Cf32) Cf32[(size_t)row * N + col] = v + (bias ? bias[col] : 0.0f);
        else      Cf16[(size_t)row * N + col] = (f16_t)v;
      }
}

// ---------------- attention scores + softmax ----------------
__global__ void __launch_bounds__(128) attn_scores_softmax(
    const f16_t* __restrict__ QKVh, f16_t* __restrict__ Ph) {
  const int blk = blockIdx.x;
  const int it = blk % 13;
  const int bh = blk / 13;
  const int b = bh >> 4, h = bh & 15;
  const int tid = threadIdx.x, lane = tid & 31, w = tid >> 5;
  __shared__ float sS[16][NPL_];
  __shared__ float red[16][8];

  for (int idx = tid; idx < 16 * NPL_; idx += 128) ((float*)sS)[idx] = -1e30f;

  const int qrow = it * 16 + (lane & 15);
  const bool rok = qrow < N_;
  const f16_t* qp =
      QKVh + (size_t)(b * N_ + (rok ? qrow : 0)) * (3 * C_) + h * DH_;
  v16h aq[2];
#pragma unroll
  for (int kb = 0; kb < 2; ++kb) {
    int koff = kb * 32 + ((lane >> 4) << 3);
    HV t{};
    if (rok) {
      t.q[0] = *(const uint4*)(qp + koff);
      t.q[1] = *(const uint4*)(qp + koff + 16);
    }
    aq[kb] = t.v;
  }
  __syncthreads();

  for (int jt = w; jt < 13; jt += 4) {
    const int j = jt * 16 + (lane & 15);
    const bool jok = j < N_;
    const f16_t* kp =
        QKVh + (size_t)(b * N_ + (jok ? j : 0)) * (3 * C_) + C_ + h * DH_;
    v8f acc = {};
#pragma unroll
    for (int kb = 0; kb < 2; ++kb) {
      int koff = kb * 32 + ((lane >> 4) << 3);
      HV t{};
      if (jok) {
        t.q[0] = *(const uint4*)(kp + koff);
        t.q[1] = *(const uint4*)(kp + koff + 16);
      }
      acc = wmma_f16(aq[kb], t.v, acc);
    }
#pragma unroll
    for (int r = 0; r < 8; ++r) {
      int m = r + ((lane >> 4) << 3);
      int jj = jt * 16 + (lane & 15);
      sS[m][jj] = (jj < N_) ? acc[r] * 0.125f : -1e30f;  // scale = Dh^-0.5
    }
  }
  __syncthreads();

  const int row = tid >> 3, l8 = tid & 7;
  float mx = -1e30f;
  for (int j = l8; j < NPL_; j += 8) mx = fmaxf(mx, sS[row][j]);
  red[row][l8] = mx;
  __syncthreads();
  if (l8 == 0) {
    float m2 = red[row][0];
#pragma unroll
    for (int q = 1; q < 8; ++q) m2 = fmaxf(m2, red[row][q]);
    red[row][0] = m2;
  }
  __syncthreads();
  mx = red[row][0];
  float s = 0.0f;
  for (int j = l8; j < NPL_; j += 8) s += __expf(sS[row][j] - mx);
  __syncthreads();
  red[row][l8] = s;
  __syncthreads();
  if (l8 == 0) {
    float t = 0.0f;
#pragma unroll
    for (int q = 0; q < 8; ++q) t += red[row][q];
    red[row][0] = t;
  }
  __syncthreads();
  float inv = 1.0f / red[row][0];
  int gi = it * 16 + row;
  if (gi < N_) {
    for (int j = l8; j < NP_; j += 8)
      Ph[((size_t)bh * N_ + gi) * NP_ + j] = (f16_t)(__expf(sS[row][j] - mx) * inv);
  }
}

// ---------------- column mean over queries ----------------
__global__ void col_mean(const f16_t* __restrict__ Ph, float* __restrict__ patch) {
  int bh = blockIdx.x, tid = threadIdx.x;
  if (tid < N_ - 1) {
    int j = tid + 1;
    float s = 0.0f;
    for (int i = 0; i < N_; ++i) s += (float)Ph[((size_t)bh * N_ + i) * NP_ + j];
    patch[bh * (N_ - 1) + tid] = s * (1.0f / (float)N_);
  }
}

// ---------------- UCB + stable top-k selection ----------------
__global__ void zero_delta(float* delta) {
  if (threadIdx.x < N_) delta[threadIdx.x] = 0.0f;
}

__global__ void ucb_select(const float* __restrict__ patch,
                           const float* __restrict__ ucb_cnt,
                           const int* __restrict__ counter,
                           float* __restrict__ keep, float* __restrict__ delta) {
  int b = blockIdx.x, tid = threadIdx.x;
  __shared__ float s[N_ - 1];
  float logc = logf((float)(*counter) + 1.0f);
  if (tid < N_ - 1) {
    float acc = 0.0f;
    for (int h = 0; h < H_; ++h) {
      float expl = sqrtf(logc / (ucb_cnt[h * N_ + tid + 1] + 1e-6f));  // BETA=1
      acc += patch[(b * H_ + h) * (N_ - 1) + tid] + expl;
    }
    s[tid] = acc * (1.0f / (float)H_);
  }
  __syncthreads();
  if (tid < N_ - 1) {
    float v = s[tid];
    int cnt = 0;
    for (int l = 0; l < N_ - 1; ++l) {
      float fl = s[l];
      cnt += (fl > v) || (fl == v && l < tid);  // stable rank (lax.top_k ties)
    }
    int sel = cnt < KKEEP;
    keep[b * N_ + tid + 1] = sel ? 1.0f : 0.0f;
    if (sel) atomicAdd(&delta[tid + 1], 1.0f / (float)B_);  // exact 1/32 steps
  }
  if (tid == N_ - 1) keep[b * N_] = 1.0f;  // CLS always kept
}

// ---------------- prune + renormalize + context (WMMA) ----------------
__global__ void __launch_bounds__(128) prune_context(
    const f16_t* __restrict__ Ph, const f16_t* __restrict__ vT,
    const float* __restrict__ keep, f16_t* __restrict__ ctxh) {
  const int blk = blockIdx.x;
  const int it = blk % 13;
  const int bh = blk / 13;
  const int b = bh >> 4, h = bh & 15;
  const int tid = threadIdx.x, lane = tid & 31, w = tid >> 5;
  __shared__ float sP[16][NPL_];
  __shared__ f16_t sPh[16][NPL_];
  __shared__ float red[16][8];

  for (int idx = tid; idx < 16 * NPL_; idx += 128) {
    int m = idx / NPL_, j = idx % NPL_;
    int gi = it * 16 + m;
    float v = 0.0f;
    if (gi < N_ && j < N_) {
      v = (float)Ph[((size_t)bh * N_ + gi) * NP_ + j];
      v *= fmaxf(keep[b * N_ + gi], keep[b * N_ + j]);  // row OR col kept
    }
    ((float*)sP)[idx] = v;
  }
  __syncthreads();

  const int row = tid >> 3, l8 = tid & 7;
  float s = 0.0f;
  for (int j = l8; j < NPL_; j += 8) s += sP[row][j];
  red[row][l8] = s;
  __syncthreads();
  if (l8 == 0) {
    float t = 0.0f;
#pragma unroll
    for (int q = 0; q < 8; ++q) t += red[row][q];
    red[row][0] = t;
  }
  __syncthreads();
  float inv = 1.0f / (red[row][0] + 1e-8f);
  for (int j = l8; j < NPL_; j += 8) sPh[row][j] = (f16_t)(sP[row][j] * inv);
  __syncthreads();

  const f16_t* vrow = vT + (size_t)(bh * DH_ + w * 16 + (lane & 15)) * NPL_;
  v8f acc = {};
  for (int kb = 0; kb < 7; ++kb) {
    const f16_t* ap = &sPh[lane & 15][kb * 32 + ((lane >> 4) << 3)];
    HV ta;
    ta.q[0] = *(const uint4*)ap;
    ta.q[1] = *(const uint4*)(ap + 16);
    const f16_t* bp = vrow + kb * 32 + ((lane >> 4) << 3);
    HV tb;
    tb.q[0] = *(const uint4*)bp;
    tb.q[1] = *(const uint4*)(bp + 16);
    acc = wmma_f16(ta.v, tb.v, acc);
  }
#pragma unroll
  for (int r = 0; r < 8; ++r) {
    int gi = it * 16 + r + ((lane >> 4) << 3);
    int col = h * DH_ + w * 16 + (lane & 15);
    if (gi < N_) ctxh[(size_t)(b * N_ + gi) * C_ + col] = (f16_t)acc[r];
  }
}

// ---------------- score_delta broadcast ----------------
__global__ void bcast_delta(const float* __restrict__ delta, float* __restrict__ outd) {
  int h = blockIdx.x;
  if (threadIdx.x < N_) outd[h * N_ + threadIdx.x] = delta[threadIdx.x];
}

// ---------------- launch ----------------
extern "C" void kernel_launch(void* const* d_in, const int* in_sizes, int n_in,
                              void* d_out, int out_size, void* d_ws, size_t ws_size,
                              hipStream_t stream) {
  const float* x       = (const float*)d_in[0];  // [32,197,1024]
  const float* Wqkv    = (const float*)d_in[1];  // [1024,3072]
  const float* Wproj   = (const float*)d_in[2];  // [1024,1024]
  const float* bproj   = (const float*)d_in[3];  // [1024]
  const float* ucb_cnt = (const float*)d_in[4];  // [16,197]
  const int*   counter = (const int*)d_in[5];    // scalar
  float* out = (float*)d_out;                    // [32,197,1024] ++ [16,197]
  (void)in_sizes; (void)n_in; (void)out_size; (void)ws_size;

  char* ws = (char*)d_ws;
  size_t off = 0;
  auto take = [&](size_t bytes) -> char* {
    char* p = ws + off;
    off += (bytes + 255) & ~(size_t)255;
    return p;
  };
  f16_t* Xh    = (f16_t*)take((size_t)M_ * C_ * 2);             // 12.9 MB
  f16_t* WqhT  = (f16_t*)take((size_t)3 * C_ * C_ * 2);         //  6.3 MB [3072][1024]
  f16_t* WphT  = (f16_t*)take((size_t)C_ * C_ * 2);             //  2.1 MB [1024][1024]
  f16_t* QKVh  = (f16_t*)take((size_t)M_ * 3 * C_ * 2);         // 38.7 MB
  f16_t* Ph    = (f16_t*)take((size_t)B_ * H_ * N_ * NP_ * 2);  // 42.0 MB
  f16_t* vT    = (f16_t*)take((size_t)B_ * H_ * DH_ * NPL_ * 2);// 14.7 MB
  float* patch = (float*)take((size_t)B_ * H_ * (N_ - 1) * 4);
  float* keep  = (float*)take((size_t)B_ * N_ * 4);
  float* delta = (float*)take((size_t)N_ * 4);
  f16_t* ctxh  = (f16_t*)take((size_t)M_ * C_ * 2);             // 12.9 MB

  // operand conversion: X row-major; weights converted WITH transpose
  cvt_f32_f16<<<(M_ * C_ / 4 + 255) / 256, 256, 0, stream>>>(x, Xh, M_ * C_ / 4);
  cvt_transpose<<<dim3(3 * C_ / 32, C_ / 32), 256, 0, stream>>>(Wqkv, WqhT, C_, 3 * C_);
  cvt_transpose<<<dim3(C_ / 32, C_ / 32), 256, 0, stream>>>(Wproj, WphT, C_, C_);
  zero_delta<<<1, 256, 0, stream>>>(delta);

  // QKV projection: [6304,1024] x [1024,3072] -> f16 (TDM double-buffered)
  gemm_wmma<<<dim3(3 * C_ / 256, M_ / 32), 256, 0, stream>>>(
      Xh, WqhT, QKVh, nullptr, nullptr, M_, 3 * C_, C_);

  // V transpose for the context GEMM's B fragments
  transpose_v<<<dim3(NPL_ / 32, DH_ / 32, B_ * H_), 256, 0, stream>>>(QKVh, vT);

  // attention probs (f16, padded) + column means + UCB selection
  attn_scores_softmax<<<B_ * H_ * 13, 128, 0, stream>>>(QKVh, Ph);
  col_mean<<<B_ * H_, 256, 0, stream>>>(Ph, patch);
  ucb_select<<<B_, 256, 0, stream>>>(patch, ucb_cnt, counter, keep, delta);

  // masked renormalized probs x V -> context f16
  prune_context<<<B_ * H_ * 13, 128, 0, stream>>>(Ph, vT, keep, ctxh);

  // output projection: [6304,1024] x [1024,1024] + bias -> fp32 d_out
  gemm_wmma<<<dim3(C_ / 256, M_ / 32), 256, 0, stream>>>(
      ctxh, WphT, nullptr, out, bproj, M_, C_, C_);

  // score_delta broadcast to all heads
  bcast_delta<<<H_, 256, 0, stream>>>(delta, out + (size_t)M_ * C_);
}
